// Encoder_3676492005396
// MI455X (gfx1250) — matrix operand order repaired and checked by reading, not compiled, not run on hardware
//
#include <hip/hip_runtime.h>

#define BATCH 64
#define SEQT  512
#define EMBD  512
#define UNITS 1024
#define G3    3072   // 3*UNITS

typedef __bf16 bf16_t;
typedef __bf16 bf16x16 __attribute__((ext_vector_type(16)));
typedef float  f32x8   __attribute__((ext_vector_type(8)));

union FragU { bf16x16 v; uint4 q[2]; };
union Pack4 { bf16_t b[4]; uint2 u; };

__device__ __forceinline__ f32x8 wmma_bf16(bf16x16 a, bf16x16 b, f32x8 c) {
  return __builtin_amdgcn_wmma_f32_16x16x32_bf16(false, a, false, b, (short)0, c, false, false);
}

// A fragment (16x32 bf16 tile of rows) from LDS.
// lanes 0-15: row M=lane, elems[0..7]=K k0..k0+7, elems[8..15]=K k0+16..k0+23, k0=kb*32
// lanes 16-31: same rows M=lane-16, K offset +8
__device__ __forceinline__ bf16x16 load_afrag(const bf16_t* sh, int row, int strideElems,
                                              int kb, int lane) {
  int k0 = kb * 32 + ((lane >> 4) << 3);
  const bf16_t* p = sh + row * strideElems + k0;
  FragU f;
  f.q[0] = *(const uint4*)(p);
  f.q[1] = *(const uint4*)(p + 16);
  return f.v;
}

// B fragment (32x16 bf16) from pre-packed weights: fragment = 32 lanes x 16 bf16 contiguous.
__device__ __forceinline__ bf16x16 load_bfrag(const bf16_t* Wp, int jfull, int kb, int KB,
                                              int lane) {
  const bf16_t* p = Wp + (((long long)(jfull * KB + kb)) * 32 + lane) * 16;
  FragU f;
  f.q[0] = *(const uint4*)(p);
  f.q[1] = *(const uint4*)(p + 8);
  return f.v;
}

// Pack W[K x 3072] (row-major f32) into WMMA-B bf16 fragments.
// dst[((j*(K/32)+kb)*32 + lane)*16 + e] = W[kb*32 + (lane>=16?8:0) + (e<8 ? e : e+8)][j*16 + (lane&15)]
__global__ __launch_bounds__(256) void pack_w(const float* __restrict__ W,
                                              bf16_t* __restrict__ dst, int K) {
  int i = blockIdx.x * 256 + threadIdx.x;
  int KB = K / 32;
  int total = 192 * KB * 512;
  if (i >= total) return;
  int e    = i & 15;
  int lane = (i >> 4) & 31;
  int rest = i >> 9;
  int kb   = rest % KB;
  int j    = rest / KB;
  int k = kb * 32 + ((lane & 16) ? 8 : 0) + (e < 8 ? e : e + 8);
  int n = j * 16 + (lane & 15);
  dst[i] = (bf16_t)W[(long long)k * G3 + n];
}

__global__ __launch_bounds__(256) void init_h(const float* __restrict__ h0f,
                                              const float* __restrict__ h0b,
                                              float* __restrict__ hAf, float* __restrict__ hAb) {
  int i = blockIdx.x * 256 + threadIdx.x;
  if (i < BATCH * UNITS) { hAf[i] = h0f[i]; hAb[i] = h0b[i]; }
}

// xg = emb[tokens] @ Wx + b[0], time-major [T*B][3072], bf16, both directions.
// grid.x = 2048 M-tiles (16 rows each, fixed t), grid.y = 16 column groups.
// Each wave: 3 consecutive j-tiles, 3 independent accumulator chains sharing A frags.
__global__ __launch_bounds__(256) void xg_kernel(const int* __restrict__ enc,
                                                 const float* __restrict__ emb,
                                                 const bf16_t* __restrict__ WxPf,
                                                 const bf16_t* __restrict__ WxPb,
                                                 const float* __restrict__ bf,
                                                 const float* __restrict__ bb,
                                                 bf16_t* __restrict__ xgf,
                                                 bf16_t* __restrict__ xgb) {
  const int STR = EMBD + 8;
  __shared__ __align__(16) bf16_t sh[16 * (EMBD + 8)];
  int tid = threadIdx.x;
  int mt  = blockIdx.x;           // M-tile: M0 = mt*16 ; M = t*64 + b
  int t   = mt / 4;               // (mt*16)/64
  int b0  = (mt * 16) & 63;

  // Stage 16 embedding rows (f32 -> bf16) with float4 loads + 8B LDS stores.
  for (int i4 = tid; i4 < 16 * (EMBD / 4); i4 += 256) {
    int row = i4 >> 7;                  // EMBD/4 = 128 per row
    int c4  = (i4 & 127) << 2;
    int tok = enc[(b0 + row) * SEQT + t];
    float4 v = *(const float4*)&emb[(long long)tok * EMBD + c4];
    Pack4 p;
    p.b[0] = (bf16_t)v.x; p.b[1] = (bf16_t)v.y;
    p.b[2] = (bf16_t)v.z; p.b[3] = (bf16_t)v.w;
    *(uint2*)&sh[row * STR + c4] = p.u;
  }
  __syncthreads();

  int lane = tid & 31, w = tid >> 5;
  int rowA = lane & 15;
  int g0   = blockIdx.y * 24 + w * 3;   // multiple of 3, never straddles dir boundary (192)
  int dir  = g0 / 192;
  int j0   = g0 % 192;
  const bf16_t* Wp   = dir ? WxPb : WxPf;
  const float*  bias = dir ? bb : bf;   // row 0 of b => b[0]
  bf16_t*       xg   = dir ? xgb : xgf;

  f32x8 a0 = {}, a1 = {}, a2 = {};
  #pragma unroll
  for (int kb = 0; kb < 16; ++kb) {
    bf16x16 a = load_afrag(sh, rowA, STR, kb, lane);
    a0 = wmma_bf16(a, load_bfrag(Wp, j0 + 0, kb, 16, lane), a0);
    a1 = wmma_bf16(a, load_bfrag(Wp, j0 + 1, kb, 16, lane), a1);
    a2 = wmma_bf16(a, load_bfrag(Wp, j0 + 2, kb, 16, lane), a2);
  }

  long long Mbase = (long long)mt * 16 + ((lane >> 4) << 3);
  #pragma unroll
  for (int it = 0; it < 3; ++it) {
    f32x8 acc = (it == 0) ? a0 : (it == 1) ? a1 : a2;
    int n = (j0 + it) * 16 + (lane & 15);
    float bv = bias[n];
    #pragma unroll
    for (int r = 0; r < 8; ++r) {
      xg[(Mbase + r) * G3 + n] = (bf16_t)(acc[r] + bv);
    }
  }
}

// One GRU timestep, both directions. 64 blocks: dir(2) x rowblock(4) x jgroup(8).
// Each wave computes one 16x16 h-tile: 3 gates x K=1024 => 96 WMMAs (3 indep chains).
__global__ __launch_bounds__(256) void gru_step(int s,
    const int* __restrict__ enc,
    const bf16_t* __restrict__ xgf, const bf16_t* __restrict__ xgb,
    const bf16_t* __restrict__ WhPf, const bf16_t* __restrict__ WhPb,
    const float* __restrict__ bf, const float* __restrict__ bb,
    const float* __restrict__ hprev_f, float* __restrict__ hnext_f,
    const float* __restrict__ hprev_b, float* __restrict__ hnext_b,
    float* __restrict__ out) {
  const int STR = UNITS + 8;
  __shared__ __align__(16) bf16_t sh[16 * (UNITS + 8)];
  int tid = threadIdx.x;
  int bx  = blockIdx.x;
  int dir = bx >> 5;
  int rb  = (bx >> 3) & 3;
  int jg  = bx & 7;
  int t   = dir ? (SEQT - 1 - s) : s;
  int tp  = dir ? (t + 1) : (t - 1);

  const float*  hprev = dir ? hprev_b : hprev_f;
  float*        hnext = dir ? hnext_b : hnext_f;
  const bf16_t* xg    = dir ? xgb : xgf;
  const bf16_t* WhP   = dir ? WhPb : WhPf;
  const float*  b1    = (dir ? bb : bf) + G3;   // b[1]

  // Stage 16 rows of h_prev (f32 -> bf16) with float4 loads + 8B LDS stores.
  for (int i4 = tid; i4 < 16 * (UNITS / 4); i4 += 256) {
    int row = i4 >> 8;                  // UNITS/4 = 256 per row
    int c4  = (i4 & 255) << 2;
    float4 v = *(const float4*)&hprev[(rb * 16 + row) * UNITS + c4];
    Pack4 p;
    p.b[0] = (bf16_t)v.x; p.b[1] = (bf16_t)v.y;
    p.b[2] = (bf16_t)v.z; p.b[3] = (bf16_t)v.w;
    *(uint2*)&sh[row * STR + c4] = p.u;
  }
  __syncthreads();

  int lane = tid & 31, w = tid >> 5;
  int j = jg * 8 + w;             // 0..63  (h column tile)
  int rowA = lane & 15;

  f32x8 az = {}, ar = {}, ah = {};
  #pragma unroll 4
  for (int kb = 0; kb < 32; ++kb) {
    bf16x16 a = load_afrag(sh, rowA, STR, kb, lane);
    az = wmma_bf16(a, load_bfrag(WhP, 0 * 64 + j, kb, 32, lane), az);
    ar = wmma_bf16(a, load_bfrag(WhP, 1 * 64 + j, kb, 32, lane), ar);
    ah = wmma_bf16(a, load_bfrag(WhP, 2 * 64 + j, kb, 32, lane), ah);
  }

  int n   = lane & 15;
  int col = j * 16 + n;
  float bz = b1[col], br = b1[UNITS + col], bh = b1[2 * UNITS + col];
  int rhi = (lane >> 4) * 8;
  #pragma unroll
  for (int r = 0; r < 8; ++r) {
    int brow = rb * 16 + rhi + r;
    long long xbase = ((long long)t * BATCH + brow) * G3;
    float xz = (float)xg[xbase + col];
    float xr = (float)xg[xbase + UNITS + col];
    float xh = (float)xg[xbase + 2 * UNITS + col];
    float hz = az[r] + bz, hr = ar[r] + br, hh = ah[r] + bh;
    float z  = 1.f / (1.f + __expf(-(xz + hz)));
    float rr = 1.f / (1.f + __expf(-(xr + hr)));
    float hc = tanhf(xh + rr * hh);
    float hp = hprev[brow * UNITS + col];
    float hn = z * hp + (1.f - z) * hc;
    bool  m  = enc[brow * SEQT + t] != 0;
    float hnew = m ? hn : hp;
    hnext[brow * UNITS + col] = hnew;
    float oprev = (s == 0) ? 0.f
                : out[((long long)brow * SEQT + tp) * (2 * UNITS) + dir * UNITS + col];
    float o = m ? hnew : oprev;
    out[((long long)brow * SEQT + t) * (2 * UNITS) + dir * UNITS + col] = o;
  }
}

__global__ __launch_bounds__(256) void finalize_state(const float* __restrict__ hf,
                                                      const float* __restrict__ hb,
                                                      float* __restrict__ out) {
  int i = blockIdx.x * 256 + threadIdx.x;
  if (i >= 2 * BATCH * UNITS) return;
  int dir = i >> 16;
  int r   = i & 65535;
  int b   = r >> 10;
  int u   = r & (UNITS - 1);
  const float* src = dir ? hb : hf;
  out[(long long)BATCH * SEQT * (2 * UNITS) + b * (2 * UNITS) + dir * UNITS + u] = src[r];
}

extern "C" void kernel_launch(void* const* d_in, const int* in_sizes, int n_in,
                              void* d_out, int out_size, void* d_ws, size_t ws_size,
                              hipStream_t stream) {
  const int*   enc = (const int*)d_in[0];
  const float* h0f = (const float*)d_in[1];
  const float* h0b = (const float*)d_in[2];
  const float* emb = (const float*)d_in[3];
  const float* Wxf = (const float*)d_in[4];
  const float* Whf = (const float*)d_in[5];
  const float* bfp = (const float*)d_in[6];
  const float* Wxb = (const float*)d_in[7];
  const float* Whb = (const float*)d_in[8];
  const float* bbp = (const float*)d_in[9];
  float* out = (float*)d_out;

  char* ws = (char*)d_ws;
  size_t off = 0;
  auto alloc = [&](size_t bytes) {
    void* p = ws + off;
    off += (bytes + 255) & ~(size_t)255;
    return p;
  };
  const size_t XGB = (size_t)SEQT * BATCH * G3 * sizeof(bf16_t);   // ~201 MB each
  bf16_t* xgf  = (bf16_t*)alloc(XGB);
  bf16_t* xgb  = (bf16_t*)alloc(XGB);
  bf16_t* WxPf = (bf16_t*)alloc((size_t)EMBD  * G3 * sizeof(bf16_t));
  bf16_t* WxPb = (bf16_t*)alloc((size_t)EMBD  * G3 * sizeof(bf16_t));
  bf16_t* WhPf = (bf16_t*)alloc((size_t)UNITS * G3 * sizeof(bf16_t));
  bf16_t* WhPb = (bf16_t*)alloc((size_t)UNITS * G3 * sizeof(bf16_t));
  float*  hAf  = (float*)alloc((size_t)BATCH * UNITS * sizeof(float));
  float*  hBf  = (float*)alloc((size_t)BATCH * UNITS * sizeof(float));
  float*  hAb  = (float*)alloc((size_t)BATCH * UNITS * sizeof(float));
  float*  hBb  = (float*)alloc((size_t)BATCH * UNITS * sizeof(float));

  // 1) Pack weights to WMMA-B bf16 fragment order.
  {
    int totX = 192 * (EMBD / 32) * 512;
    int totH = 192 * (UNITS / 32) * 512;
    pack_w<<<(totX + 255) / 256, 256, 0, stream>>>(Wxf, WxPf, EMBD);
    pack_w<<<(totX + 255) / 256, 256, 0, stream>>>(Wxb, WxPb, EMBD);
    pack_w<<<(totH + 255) / 256, 256, 0, stream>>>(Whf, WhPf, UNITS);
    pack_w<<<(totH + 255) / 256, 256, 0, stream>>>(Whb, WhPb, UNITS);
  }
  init_h<<<(BATCH * UNITS + 255) / 256, 256, 0, stream>>>(h0f, h0b, hAf, hAb);

  // 2) Input projection GEMM (embedding gather fused), both directions.
  xg_kernel<<<dim3(2048, 16), 256, 0, stream>>>(enc, emb, WxPf, WxPb, bfp, bbp, xgf, xgb);

  // 3) Sequential recurrence: one launch per timestep, both directions per launch.
  for (int s = 0; s < SEQT; ++s) {
    const float* pf = (s & 1) ? hBf : hAf;
    float*       nf = (s & 1) ? hAf : hBf;
    const float* pb = (s & 1) ? hBb : hAb;
    float*       nb = (s & 1) ? hAb : hBb;
    gru_step<<<64, 256, 0, stream>>>(s, enc, xgf, xgb, WhPf, WhPb, bfp, bbp,
                                     pf, nf, pb, nb, out);
  }

  // 4) Final states (T=512 even -> final h lives in the 'A' buffers).
  finalize_state<<<(2 * BATCH * UNITS + 255) / 256, 256, 0, stream>>>(hAf, hAb, out);
}
